// BiLSTM_51780125721270
// MI455X (gfx1250) — compile-verified, matrix operand order checked by reading, not run on hardware
//
#include <hip/hip_runtime.h>

// BiLSTM for MI455X (gfx1250, wave32, WMMA + TDM).
// Phase 1: xp = x @ Wk + b  (per direction, big WMMA GEMM, f16 in / f32 acc)
// Phase 2: persistent scan, 4 WGs per direction, Wr held in VGPRs as WMMA
//          B-tiles, h exchanged via L2 + spin barrier each timestep, xp slab
//          staged into LDS by the Tensor Data Mover each step.

#define BB 32
#define TT 512
#define DD 256
#define UU 256
#define GG 1024          // 4*UU
#define NWG 4            // workgroups per direction in the scan

typedef __attribute__((ext_vector_type(16))) _Float16 v16h;
typedef __attribute__((ext_vector_type(8)))  _Float16 v8h;
typedef __attribute__((ext_vector_type(8)))  float    v8f;
typedef __attribute__((ext_vector_type(4)))  float    v4f;
typedef unsigned int u32x4 __attribute__((ext_vector_type(4)));
typedef int          i32x4 __attribute__((ext_vector_type(4)));
typedef int          i32x8 __attribute__((ext_vector_type(8)));

// workspace layout (bytes)
#define WS_SYNC_OFF 0
#define WS_HEX_OFF  1024                       // 2 dir * 2 parity * 32*256 f16 = 64KB
#define WS_XP_OFF   131072                     // 2 * T*B*G f32 = 128MB

__device__ __forceinline__ v16h pack16(v8h lo, v8h hi) {
  v16h r;
#pragma unroll
  for (int i = 0; i < 8; i++) { r[i] = lo[i]; r[i + 8] = hi[i]; }
  return r;
}

// 8 contiguous f32 from global -> 8 halfs (32B-aligned source)
__device__ __forceinline__ v8h cvt8g(const float* p) {
  v4f a = *(const v4f*)p;
  v4f b = *(const v4f*)(p + 4);
  v8h r;
#pragma unroll
  for (int i = 0; i < 4; i++) { r[i] = (_Float16)a[i]; r[i + 4] = (_Float16)b[i]; }
  return r;
}

__device__ __forceinline__ float sigmoidf_(float x) {
  return 1.0f / (1.0f + __expf(-x));
}

// Issue a TDM load of a 3D tile (dim0=64 f32 contiguous, dim1=4 @stride 256,
// dim2=32 @stride 1024 elements) from global xp into LDS (contiguous 32KB).
__device__ __forceinline__ void tdm_load_xp_slab(const float* gsrc, unsigned lds_addr) {
  unsigned long long ga = (unsigned long long)(uintptr_t)gsrc;

  u32x4 g0;
  g0[0] = 1u;                                     // count=1, user descriptor
  g0[1] = lds_addr;                               // lds_addr (bytes)
  g0[2] = (unsigned)(ga & 0xffffffffu);           // global_addr[31:0]
  g0[3] = (unsigned)((ga >> 32) & 0x01ffffffu)    // global_addr[56:32]
        | (2u << 30);                             // type = 2 ("image")

  i32x8 g1;
  g1[0] = (int)(2u << 16);                        // workgroup_mask=0, data_size=2 (4B)
  g1[1] = (int)((64u & 0xffffu) << 16);           // tensor_dim0[15:0]=64 (bits 63:48)
  g1[2] = (int)((4u & 0xffffu) << 16);            // dim0[31:16]=0 | tensor_dim1[15:0]=4
  g1[3] = (int)((64u & 0xffffu) << 16);           // dim1[31:16]=0 | tile_dim0=64
  g1[4] = (int)((4u) | (32u << 16));              // tile_dim1=4 | tile_dim2=32
  g1[5] = (int)256;                               // tensor_dim0_stride[31:0] = 256
  g1[6] = (int)((1024u & 0xffffu) << 16);         // s0[47:32]=0 | tensor_dim1_stride[15:0]=1024
  g1[7] = 0;                                      // tensor_dim1_stride[47:16] = 0

  i32x4 g2;
  g2[0] = 32;                                     // tensor_dim2 = 32
  g2[1] = 0;                                      // tensor_dim3 = 0
  g2[2] = 0;                                      // tensor_dim2_stride (unused, tile_dim3=0)
  g2[3] = 0;                                      // tile_dim3 = 0
  i32x4 g3 = {0, 0, 0, 0};

#if defined(__clang_major__) && (__clang_major__ >= 23)
  i32x8 z8 = {0, 0, 0, 0, 0, 0, 0, 0};
  __builtin_amdgcn_tensor_load_to_lds(g0, g1, g2, g3, z8, 0);
#else
  __builtin_amdgcn_tensor_load_to_lds(g0, g1, g2, g3, 0);
#endif
}

// ---------------------------------------------------------------------------
// Kernel 1: input projection xp[dir][t*B+b][4U] = x[b, t(dir)] @ Wk + bias
// grid = (G/64, (T*B)/128, 2), block = 256 (8 waves)
// ---------------------------------------------------------------------------
__global__ __launch_bounds__(256) void bilstm_proj(
    const float* __restrict__ x,
    const float* __restrict__ Wk_f, const float* __restrict__ b_f,
    const float* __restrict__ Wk_b, const float* __restrict__ b_b,
    float* __restrict__ xp, unsigned* __restrict__ syncv)
{
  const int dir = blockIdx.z;
  const float* Wk   = dir ? Wk_b : Wk_f;
  const float* bias = dir ? b_b  : b_f;
  float* xpd = xp + (size_t)dir * ((size_t)TT * BB * GG);

  // reset scan sync counters for this launch (runs before the scan kernel)
  if (blockIdx.x == 0 && blockIdx.y == 0 && blockIdx.z == 0 && threadIdx.x < 8)
    syncv[threadIdx.x] = 0u;

  __shared__ _Float16 Bs[64][DD + 8];   // Wk slice, [n_local][k], 16B-aligned rows

  const int n0  = blockIdx.x * 64;
  const int m0  = blockIdx.y * 128;
  const int tid = threadIdx.x;

  // stage Wk columns [n0, n0+64) into LDS transposed (f32 -> f16)
  {
    int n  = tid & 63;
    int k4 = tid >> 6;                  // 0..3
    for (int k = k4; k < DD; k += 4)
      Bs[n][k] = (_Float16)Wk[(size_t)k * GG + n0 + n];
  }
  __syncthreads();

  const int wave  = tid >> 5;
  const int lane  = tid & 31;
  const int mw    = (wave >> 1) * 2;    // M-tile base: 0,2,4,6
  const int nw    = (wave & 1) * 2;     // N-tile base: 0,2
  const int l16   = lane & 15;
  const int lhalf = (lane < 16) ? 0 : 8;

  // accumulators start at the bias (WMMA computes A*B + C)
  v8f acc[2][2];
#pragma unroll
  for (int jn = 0; jn < 2; jn++) {
    float bv = bias[n0 + (nw + jn) * 16 + l16];
#pragma unroll
    for (int im = 0; im < 2; im++)
#pragma unroll
      for (int r = 0; r < 8; r++) acc[im][jn][r] = bv;
  }

  // per-lane x row pointers (xp row = t*B + b; x row = b*T + t, time-reversed for bw)
  const float* arow[2];
#pragma unroll
  for (int im = 0; im < 2; im++) {
    int r    = m0 + (mw + im) * 16 + l16;
    int t    = r >> 5;
    int bidx = r & 31;
    int st   = dir ? (TT - 1 - t) : t;
    arow[im] = x + ((size_t)bidx * TT + st) * DD;
  }

#pragma unroll 2
  for (int kt = 0; kt < 8; kt++) {
    int k0 = kt * 32 + lhalf;
    v16h a[2], bm[2];
#pragma unroll
    for (int im = 0; im < 2; im++)
      a[im] = pack16(cvt8g(arow[im] + k0), cvt8g(arow[im] + k0 + 16));
#pragma unroll
    for (int jn = 0; jn < 2; jn++) {
      const _Float16* bp = &Bs[(nw + jn) * 16 + l16][k0];
      bm[jn] = pack16(*(const v8h*)bp, *(const v8h*)(bp + 16));
    }
#pragma unroll
    for (int im = 0; im < 2; im++)
#pragma unroll
      for (int jn = 0; jn < 2; jn++)
        acc[im][jn] = __builtin_amdgcn_wmma_f32_16x16x32_f16(
            false, a[im], false, bm[jn], (short)0, acc[im][jn], false, false);
  }

  // write xp (f32); keep default temporal policy -> xp stays hot in the 192MB L2
#pragma unroll
  for (int im = 0; im < 2; im++)
#pragma unroll
    for (int jn = 0; jn < 2; jn++) {
      int Ncol = n0 + (nw + jn) * 16 + l16;
#pragma unroll
      for (int r = 0; r < 8; r++) {
        int Mrow = m0 + (mw + im) * 16 + r + lhalf;
        xpd[(size_t)Mrow * GG + Ncol] = acc[im][jn][r];
      }
    }
}

// ---------------------------------------------------------------------------
// Kernel 2: persistent recurrent scan. grid = (NWG, 2), block = 512 (16 waves)
// WG (j, dir) owns units [j*64, j*64+64) of direction dir across all 4 gates.
// Wr B-tiles live in VGPRs for the whole kernel; h crosses WGPs via L2; the
// per-step xp slab is DMA'd into LDS by the TDM under the recurrent GEMM.
// ---------------------------------------------------------------------------
__global__ __launch_bounds__(512) void bilstm_scan(
    const float* __restrict__ Wr_f, const float* __restrict__ Wr_b,
    const float* __restrict__ xp, _Float16* __restrict__ hex,
    float* __restrict__ out, unsigned* __restrict__ syncv)
{
  const int dir = blockIdx.y;
  const int j   = blockIdx.x;                 // unit-slice 0..3
  const float* Wr  = dir ? Wr_b : Wr_f;
  const float* xpd = xp + (size_t)dir * ((size_t)TT * BB * GG);
  _Float16* hx0 = hex + (size_t)(dir * 2 + 0) * (BB * UU);
  _Float16* hx1 = hex + (size_t)(dir * 2 + 1) * (BB * UU);
  unsigned* cnt = syncv + dir;

  __shared__ float zbuf[BB][4 * 64 + 4];      // gate-major pre-activations
  __shared__ float xbuf[BB][4 * 64];          // TDM-staged xp slab: [b][g*64+lu]

  const int tid   = threadIdx.x;
  const int wave  = tid >> 5;
  const int lane  = tid & 31;
  const int g     = wave >> 2;                // gate 0..3 (i,f,c,o)
  const int lt    = wave & 3;                 // 16-col tile inside 64-unit slice
  const int l16   = lane & 15;
  const int lhalf = (lane < 16) ? 0 : 8;
  const int Ng    = g * UU + j * 64 + lt * 16 + l16;   // this lane's Wr column

  const unsigned xbuf_lds = (unsigned)(uintptr_t)(&xbuf[0][0]);

  // --- load persistent Wr B-tiles into registers (once) ---
  v16h Wb[8];
#pragma unroll
  for (int kt = 0; kt < 8; kt++) {
    int k0 = kt * 32 + lhalf;
    v8h lo, hi;
#pragma unroll
    for (int i = 0; i < 8; i++) {
      lo[i] = (_Float16)Wr[(size_t)(k0 + i)      * GG + Ng];
      hi[i] = (_Float16)Wr[(size_t)(k0 + 16 + i) * GG + Ng];
    }
    Wb[kt] = pack16(lo, hi);
  }

  // elementwise ownership: 4 (batch, unit) pairs per thread, c state in regs
  int bidx[4], lu[4];
  float cst[4] = {0.f, 0.f, 0.f, 0.f};
#pragma unroll
  for (int p = 0; p < 4; p++) {
    int idx = p * 512 + tid;
    bidx[p] = idx >> 6;
    lu[p]   = idx & 63;
  }

  // zero this WG's slice of both h parity buffers (h(-1) = 0)
  for (int q = tid; q < 2 * BB * 64; q += 512) {
    int buf = (q >= BB * 64);
    int rem = q - buf * BB * 64;
    int b   = rem >> 6;
    int u   = j * 64 + (rem & 63);
    (buf ? hx1 : hx0)[b * UU + u] = (_Float16)0.0f;
  }
  __threadfence();
  __syncthreads();
  if (tid == 0) {
    __hip_atomic_fetch_add(cnt, 1u, __ATOMIC_ACQ_REL, __HIP_MEMORY_SCOPE_AGENT);
    while (__hip_atomic_load(cnt, __ATOMIC_ACQUIRE, __HIP_MEMORY_SCOPE_AGENT) < (unsigned)NWG)
      __builtin_amdgcn_s_sleep(1);
  }
  __syncthreads();

  for (int t = 0; t < TT; t++) {
    const _Float16* hin = (t & 1) ? hx1 : hx0;
    _Float16* hout      = (t & 1) ? hx0 : hx1;

    // kick the TDM for this step's xp slab; the GEMM below hides its latency
    if (wave == 0 && lane == 0)
      tdm_load_xp_slab(xpd + ((size_t)t * BB) * GG + j * 64, xbuf_lds);
    // warm L2 for the next step's slab
    if (tid == 32 && t + 1 < TT)
      __builtin_prefetch(xpd + ((size_t)(t + 1) * BB) * GG + j * 64, 0, 0);

    // z-slice = h(t-1) @ Wr[:, owned columns]
    v8f acc[2];
#pragma unroll
    for (int im = 0; im < 2; im++)
#pragma unroll
      for (int r = 0; r < 8; r++) acc[im][r] = 0.0f;

#pragma unroll
    for (int kt = 0; kt < 8; kt++) {
      int k0 = kt * 32 + lhalf;
#pragma unroll
      for (int im = 0; im < 2; im++) {
        const _Float16* ap = hin + (im * 16 + l16) * UU + k0;
        v16h a = pack16(*(const v8h*)ap, *(const v8h*)(ap + 16));
        acc[im] = __builtin_amdgcn_wmma_f32_16x16x32_f16(
            false, a, false, Wb[kt], (short)0, acc[im], false, false);
      }
    }

    // dump z tiles to LDS (C layout: VGPR r -> M = r + lane-half*8, N = lane%16)
    {
      int col = g * 64 + lt * 16 + l16;
#pragma unroll
      for (int im = 0; im < 2; im++)
#pragma unroll
        for (int r = 0; r < 8; r++)
          zbuf[im * 16 + r + lhalf][col] = acc[im][r];
    }
    // drain the TDM before anyone reads xbuf; barrier publishes it WG-wide
    if (wave == 0 && lane == 0)
      __builtin_amdgcn_s_wait_tensorcnt(0);
    __syncthreads();

    // gates + state update + outputs
#pragma unroll
    for (int p = 0; p < 4; p++) {
      int b = bidx[p];
      int u = j * 64 + lu[p];
      float zi = zbuf[b][0 * 64 + lu[p]] + xbuf[b][0 * 64 + lu[p]];
      float zf = zbuf[b][1 * 64 + lu[p]] + xbuf[b][1 * 64 + lu[p]];
      float zg = zbuf[b][2 * 64 + lu[p]] + xbuf[b][2 * 64 + lu[p]];
      float zo = zbuf[b][3 * 64 + lu[p]] + xbuf[b][3 * 64 + lu[p]];
      float ig = sigmoidf_(zi);
      float fg = sigmoidf_(zf);
      float gv = tanhf(zg);
      float og = sigmoidf_(zo);
      cst[p] = fg * cst[p] + ig * gv;
      float h = og * tanhf(cst[p]);

      hout[b * UU + u] = (_Float16)h;
      int tout = dir ? (TT - 1 - t) : t;
      // write-once outputs: non-temporal so they don't rinse xp out of L2
      __builtin_nontemporal_store(h, &out[((size_t)b * TT + tout) * (2 * UU) + dir * UU + u]);
      if (t == TT - 1)
        __builtin_nontemporal_store(
            h, &out[(size_t)BB * TT * (2 * UU) + (size_t)b * (2 * UU) + dir * UU + u]);
    }

    // publish h(t), then cross-WG barrier for this direction
    __threadfence();
    __syncthreads();
    if (tid == 0) {
      __hip_atomic_fetch_add(cnt, 1u, __ATOMIC_ACQ_REL, __HIP_MEMORY_SCOPE_AGENT);
      unsigned tgt = (unsigned)NWG * (unsigned)(t + 2);
      while (__hip_atomic_load(cnt, __ATOMIC_ACQUIRE, __HIP_MEMORY_SCOPE_AGENT) < tgt)
        __builtin_amdgcn_s_sleep(1);
    }
    __syncthreads();
  }
}

// ---------------------------------------------------------------------------
extern "C" void kernel_launch(void* const* d_in, const int* in_sizes, int n_in,
                              void* d_out, int out_size, void* d_ws, size_t ws_size,
                              hipStream_t stream)
{
  const float* x    = (const float*)d_in[0];
  const float* Wk_f = (const float*)d_in[1];
  const float* Wr_f = (const float*)d_in[2];
  const float* b_f  = (const float*)d_in[3];
  const float* Wk_b = (const float*)d_in[4];
  const float* Wr_b = (const float*)d_in[5];
  const float* b_b  = (const float*)d_in[6];
  float* out = (float*)d_out;

  unsigned*  syncv = (unsigned*)((char*)d_ws + WS_SYNC_OFF);
  _Float16*  hex   = (_Float16*)((char*)d_ws + WS_HEX_OFF);
  float*     xp    = (float*)((char*)d_ws + WS_XP_OFF);

  dim3 gp(GG / 64, (TT * BB) / 128, 2);
  bilstm_proj<<<gp, 256, 0, stream>>>(x, Wk_f, b_f, Wk_b, b_b, xp, syncv);

  dim3 gs(NWG, 2);
  bilstm_scan<<<gs, 512, 0, stream>>>(Wr_f, Wr_b, xp, hex, out, syncv);
}
